// ProxyISA_10222022165009
// MI455X (gfx1250) — matrix operand order repaired and checked by reading, not compiled, not run on hardware
//
#include <hip/hip_runtime.h>
#include <hip/hip_bf16.h>
#include <math.h>

typedef __attribute__((ext_vector_type(16))) _Float16 v16h;
typedef __attribute__((ext_vector_type(8)))  _Float16 v8h;
typedef __attribute__((ext_vector_type(8)))  float    v8f;

#define Bsz   1024
#define Esz   512
#define Csz   16384
#define MRG   0.1f
#define ALPHA 32.0f
#define KCOEF 1.0f
#define LAMB  0.1f

#define WG_CLASSES 64
#define WG_THREADS 128          // 4 waves of 32
#define N_WAVES    4
#define ROW_F16    (Esz + 8)    // padded LDS row stride (f16 elems); 1040 B, 16B aligned

union AB16 { v16h v; v8h h[2]; };
union Acc8 { v8f v; float f[8]; };

// ---------------------------------------------------------------------------
// Kernel 1: l2-normalize proxies -> f16, precompute per-class scalars.
// One wave (32 lanes) per class row of 512 floats.
// ---------------------------------------------------------------------------
__global__ __launch_bounds__(32)
void prep_proxies(const float* __restrict__ proxies,
                  const float* __restrict__ effNum,
                  const float* __restrict__ lsim,
                  _Float16* __restrict__ Pn,
                  float* __restrict__ outlier,
                  float* __restrict__ invEff) {
  const int c    = blockIdx.x;
  const int lane = threadIdx.x;
  const float* row = proxies + ((size_t)c << 9);
  float vals[16];
  float ss = 0.0f;
#pragma unroll
  for (int i = 0; i < 16; ++i) {
    float x = row[lane + i * 32];
    vals[i] = x;
    ss += x * x;
  }
#pragma unroll
  for (int off = 16; off >= 1; off >>= 1) ss += __shfl_xor(ss, off, 32);
  const float rn = rsqrtf(ss + 1e-12f);
#pragma unroll
  for (int i = 0; i < 16; ++i)
    Pn[((size_t)c << 9) + lane + i * 32] = (_Float16)(vals[i] * rn);

  if (lane == 0) {
    float en = effNum[c], ls = lsim[c];
    float w_base = 1.0f / (1.0f + log1pf(en));
    float eta    = (1.0f + KCOEF * (1.0f - ls)) * w_base + LAMB;
    outlier[c]   = ls - eta;
    invEff[c]    = 1.0f / fmaxf(1.0f, en);
  }
}

// ---------------------------------------------------------------------------
// Kernel 2: X f32 -> f16
// ---------------------------------------------------------------------------
__global__ __launch_bounds__(256)
void conv_x(const float* __restrict__ X, _Float16* __restrict__ Xh, int n) {
  int i = blockIdx.x * blockDim.x + threadIdx.x;
  if (i < n) Xh[i] = (_Float16)X[i];
}

// ---------------------------------------------------------------------------
// Kernel 3: fused GEMM (v_wmma_f32_16x16x32_f16) + ProxyAnchor epilogue.
// Block = 128 threads (4 waves), owns 64 classes.
// Proxy tile is staged into LDS with GLOBAL_LOAD_ASYNC_TO_LDS_B128 (ASYNCcnt).
// Wave w handles classes [classBase + 16w, classBase + 16w + 16).
// ---------------------------------------------------------------------------
__global__ __launch_bounds__(WG_THREADS)
void proxy_main(const _Float16* __restrict__ Xh,
                const _Float16* __restrict__ Pn,
                const int* __restrict__ targets,
                const float* __restrict__ outlier,
                const float* __restrict__ invEff,
                float4* __restrict__ partials) {
  extern __shared__ _Float16 lds[];   // [WG_CLASSES][ROW_F16], at LDS offset 0

  const int tid       = threadIdx.x;
  const int lane      = tid & 31;
  const int wave      = tid >> 5;
  const int classBase = blockIdx.x * WG_CLASSES;
  const int nlo       = lane & 15;    // column / row-within-tile selector
  const int hi        = lane >> 4;    // 0: lanes 0-15, 1: lanes 16-31

  // ---- Async-stage proxy tile into LDS: 64 rows x 512 f16, 16B per lane-op.
  // Dynamic LDS is the only LDS object -> our lds[] indices are absolute LDS
  // byte offsets, matching the VDST operand of the async copy.
  for (int ch = tid; ch < WG_CLASSES * (Esz / 8); ch += WG_THREADS) {
    int row = ch >> 6;               // 64 chunks of 8 f16 per row
    int col = (ch & 63) << 3;
    unsigned ldsOff = (unsigned)((row * ROW_F16 + col) * sizeof(_Float16));
    const _Float16* src = &Pn[((size_t)(classBase + row) << 9) + col];
    asm volatile("global_load_async_to_lds_b128 %0, %1, off"
                 :: "v"(ldsOff), "v"(src)
                 : "memory");
  }
  asm volatile("s_wait_asynccnt 0x0" ::: "memory");
  __syncthreads();

  const int   myClass = classBase + wave * 16 + nlo;
  const float osim    = outlier[myClass];
  const float ivE     = invEff[myClass];

  // B-matrix lane view: N = lane&15, K-subblocks {0..7,16..23} or {8..15,24..31}
  const _Float16* brow  = &lds[(wave * 16 + nlo) * ROW_F16];
  const int       bkoff = hi * 8;
  // A-matrix lane view: M = lane&15, K-subblocks {0..7,16..23} or {8..15,24..31}
  const int       akoff = hi * 8;
  // C/D layout: VGPR v of this lane is row (v + 8*hi), col nlo.
  const int       rowOff = hi * 8;

  float posSum = 0.0f, posCnt = 0.0f, negValSum = 0.0f, negSum = 0.0f;
  const v8f zero = {0.f, 0.f, 0.f, 0.f, 0.f, 0.f, 0.f, 0.f};

  for (int bt = 0; bt < Bsz / 16; ++bt) {
    const _Float16* arow = Xh + ((size_t)(bt * 16 + nlo) << 9);

    // targets for this lane's 8 C/D rows: two b128 loads, issued early.
    int4 tA = *(const int4*)&targets[bt * 16 + rowOff];
    int4 tB = *(const int4*)&targets[bt * 16 + rowOff + 4];

    Acc8 cacc;
    cacc.v = zero;

    // ---- software-pipelined K loop: fetch k+1 while WMMA consumes k ----
    AB16 a, b;
    a.h[0] = *(const v8h*)&arow[akoff];
    a.h[1] = *(const v8h*)&arow[16 + akoff];
    b.h[0] = *(const v8h*)&brow[bkoff];
    b.h[1] = *(const v8h*)&brow[bkoff + 16];
#pragma unroll
    for (int k0 = 32; k0 < Esz; k0 += 32) {
      AB16 an, bn;
      an.h[0] = *(const v8h*)&arow[k0 + akoff];
      an.h[1] = *(const v8h*)&arow[k0 + 16 + akoff];
      bn.h[0] = *(const v8h*)&brow[k0 + bkoff];
      bn.h[1] = *(const v8h*)&brow[k0 + bkoff + 16];
      cacc.v = __builtin_amdgcn_wmma_f32_16x16x32_f16(
          false, a.v, false, b.v, (short)0, cacc.v, false, false);
      a = an;
      b = bn;
    }
    cacc.v = __builtin_amdgcn_wmma_f32_16x16x32_f16(
        false, a.v, false, b.v, (short)0, cacc.v, false, false);

    // ---- branchless fused epilogue: exactly one exp per element ----
    int tv[8] = {tA.x, tA.y, tA.z, tA.w, tB.x, tB.y, tB.z, tB.w};
#pragma unroll
    for (int v = 0; v < 8; ++v) {
      float cv    = cacc.f[v];
      bool  isPos = (tv[v] == myClass);
      float nv    = (cv < osim) ? ivE : 1.0f;          // neg reweight
      float arg   = isPos ? (ALPHA * (MRG - cv))
                          : (ALPHA * (cv + MRG) * nv); // mask==1 at positives
      float e     = __expf(arg);
      posSum    += isPos ? e : 0.0f;
      posCnt    += isPos ? 1.0f : 0.0f;
      negValSum += isPos ? 0.0f : nv;
      negSum    += isPos ? 0.0f : e;
    }
  }

  // Lanes L and L+16 hold the same class column -> combine.
  posSum    += __shfl_xor(posSum, 16, 32);
  posCnt    += __shfl_xor(posCnt, 16, 32);
  negValSum += __shfl_xor(negValSum, 16, 32);
  negSum    += __shfl_xor(negSum, 16, 32);

  float posTerm = 0.f, negTerm = 0.f, posW = 0.f, negW = 0.f;
  if (lane < 16) {
    posTerm = log1pf(posSum);
    negTerm = log1pf(negSum);
    posW    = (posCnt > 0.0f) ? 1.0f : 0.0f;
    float Ncnt = (float)Bsz - posCnt;
    negW    = (Ncnt > 0.0f) ? (negValSum / fmaxf(Ncnt, 1.0f)) : 0.0f;
  }
#pragma unroll
  for (int off = 8; off >= 1; off >>= 1) {
    posTerm += __shfl_xor(posTerm, off, 32);
    negTerm += __shfl_xor(negTerm, off, 32);
    posW    += __shfl_xor(posW, off, 32);
    negW    += __shfl_xor(negW, off, 32);
  }

  // Cross-wave reduction: proxy tile no longer needed, reuse LDS.
  __syncthreads();
  float* red = (float*)lds;
  if (lane == 0) {
    red[wave * 4 + 0] = posTerm;
    red[wave * 4 + 1] = negTerm;
    red[wave * 4 + 2] = posW;
    red[wave * 4 + 3] = negW;
  }
  __syncthreads();
  if (tid == 0) {
    float4 p = {0.f, 0.f, 0.f, 0.f};
    for (int w = 0; w < N_WAVES; ++w) {
      p.x += red[w * 4 + 0];
      p.y += red[w * 4 + 1];
      p.z += red[w * 4 + 2];
      p.w += red[w * 4 + 3];
    }
    partials[blockIdx.x] = p;   // deterministic per-block partial
  }
}

// ---------------------------------------------------------------------------
// Kernel 4: deterministic final reduction of 256 block partials -> scalar.
// ---------------------------------------------------------------------------
__global__ void finalize(const float4* __restrict__ partials, int n,
                         float* __restrict__ out) {
  if (threadIdx.x == 0 && blockIdx.x == 0) {
    float pos = 0.f, neg = 0.f, posW = 0.f, negW = 0.f;
    for (int i = 0; i < n; ++i) {
      float4 p = partials[i];
      pos += p.x; neg += p.y; posW += p.z; negW += p.w;
    }
    out[0] = pos / posW + neg / negW;
  }
}

// ---------------------------------------------------------------------------
extern "C" void kernel_launch(void* const* d_in, const int* in_sizes, int n_in,
                              void* d_out, int out_size, void* d_ws, size_t ws_size,
                              hipStream_t stream) {
  const float* X       = (const float*)d_in[0];   // [B,E] f32
  const int*   targets = (const int*)d_in[1];     // [B]   i32
  const float* proxies = (const float*)d_in[2];   // [C,E] f32
  const float* effNum  = (const float*)d_in[3];   // [C]   f32
  const float* lsim    = (const float*)d_in[4];   // [C]   f32

  char* ws = (char*)d_ws;
  _Float16* Pn      = (_Float16*)(ws);                               // 16 MB
  _Float16* Xh      = (_Float16*)(ws + (size_t)16 * 1024 * 1024);    // 1 MB
  float*    outlier = (float*)(ws + (size_t)17 * 1024 * 1024);       // 64 KB
  float*    invEff  = (float*)(ws + (size_t)17 * 1024 * 1024 + 65536);
  float4*   parts   = (float4*)(ws + (size_t)17 * 1024 * 1024 + 2 * 65536);

  prep_proxies<<<Csz, 32, 0, stream>>>(proxies, effNum, lsim, Pn, outlier, invEff);
  conv_x<<<(Bsz * Esz + 255) / 256, 256, 0, stream>>>(X, Xh, Bsz * Esz);

  const int nBlocks = Csz / WG_CLASSES;            // 256
  size_t ldsBytes = (size_t)WG_CLASSES * ROW_F16 * sizeof(_Float16); // 66,560 B
  proxy_main<<<nBlocks, WG_THREADS, ldsBytes, stream>>>(
      Xh, Pn, targets, outlier, invEff, parts);

  finalize<<<1, 32, 0, stream>>>(parts, nBlocks, (float*)d_out);
}
